// Block5_70549132804876
// MI455X (gfx1250) — matrix operand order repaired and checked by reading, not compiled, not run on hardware
//
#include <hip/hip_runtime.h>
#include <math.h>

typedef __attribute__((ext_vector_type(16))) __bf16 v16bf;
typedef __attribute__((ext_vector_type(8)))  __bf16 v8bf;
typedef __attribute__((ext_vector_type(8)))  float  v8f;

#define TOK   50176L      // 256*49*4 tokens
#define CDIM  768
#define NH    12
#define HD    64

__device__ __forceinline__ __bf16 f2bf(float f) {
  unsigned u = __builtin_bit_cast(unsigned, f);
  unsigned short s = (unsigned short)((u + 0x7FFFu + ((u >> 16) & 1u)) >> 16);
  return __builtin_bit_cast(__bf16, s);
}
__device__ __forceinline__ float bf2f(__bf16 h) {
  unsigned short s = __builtin_bit_cast(unsigned short, h);
  unsigned u = ((unsigned)s) << 16;
  return __builtin_bit_cast(float, u);
}

// ---------------- fp32 -> bf16 convert (weights) ----------------
__global__ __launch_bounds__(256) void cvt_bf16_kernel(const float* __restrict__ in,
                                                       __bf16* __restrict__ out, long n) {
  long i = (long)blockIdx.x * 256 + threadIdx.x;
  if (i < n) out[i] = f2bf(in[i]);
}

// ---------------- LayerNorm -> bf16 ----------------
__global__ __launch_bounds__(256) void ln_bf16_kernel(const float* __restrict__ x,
                                                      const float* __restrict__ g,
                                                      const float* __restrict__ b,
                                                      __bf16* __restrict__ out) {
  __shared__ float sh_s[256], sh_q[256];
  const int tid = threadIdx.x;
  const long base = (long)blockIdx.x * CDIM;
  float s = 0.f, q = 0.f;
  for (int c = tid; c < CDIM; c += 256) { float v = x[base + c]; s += v; q += v * v; }
  sh_s[tid] = s; sh_q[tid] = q; __syncthreads();
  for (int off = 128; off > 0; off >>= 1) {
    if (tid < off) { sh_s[tid] += sh_s[tid + off]; sh_q[tid] += sh_q[tid + off]; }
    __syncthreads();
  }
  const float mu   = sh_s[0] * (1.0f / CDIM);
  const float var  = sh_q[0] * (1.0f / CDIM) - mu * mu;
  const float rstd = rsqrtf(var + 1e-5f);
  for (int c = tid; c < CDIM; c += 256)
    out[base + c] = f2bf((x[base + c] - mu) * rstd * g[c] + b[c]);
}

// ---------------- WMMA bf16 GEMM ----------------
// out[m,n] = sum_k A[m,k] * W[n,k]   (A: [M,K] bf16, W: [N,K] bf16)
// Wave tile 32x64 (2 row-frags x 4 col-frags, 8 f32 accumulators).
// Block = 8 waves arranged 4(m) x 2(n) -> 128x128 block tile.
// MODE 0: store bf16 plain        (QKV)
// MODE 1: +bias +resid, store f32 (proj / fc2)
// MODE 2: +bias, GELU, store bf16 (fc1)
template <int MODE>
__device__ __forceinline__ void store_frag(const v8f& c, long row0, long col0, int hf, int ln,
                                           const float* __restrict__ bias,
                                           const float* __restrict__ resid,
                                           float* __restrict__ outF,
                                           __bf16* __restrict__ outB, long N) {
  const long col = col0 + ln;
  float bia = 0.f;
  if (MODE == 1 || MODE == 2) bia = bias[col];
#pragma unroll
  for (int r = 0; r < 8; r++) {
    const long row = row0 + hf * 8 + r;
    float v = c[r];
    if (MODE == 1 || MODE == 2) v += bia;
    if (MODE == 2) v = 0.5f * v * (1.0f + erff(v * 0.70710678118f));  // exact GELU
    if (MODE == 1) {
      v += resid[row * N + col];
      outF[row * N + col] = v;
    } else {
      outB[row * N + col] = f2bf(v);
    }
  }
}

template <int MODE>
__global__ __launch_bounds__(256) void gemm_wmma_kernel(const __bf16* __restrict__ A,
                                                        const __bf16* __restrict__ W,
                                                        const float* __restrict__ bias,
                                                        const float* __restrict__ resid,
                                                        float* __restrict__ outF,
                                                        __bf16* __restrict__ outB,
                                                        int N, int K) {
  const int tid  = threadIdx.x;
  const int lane = tid & 31;
  const int wave = tid >> 5;
  const int ln   = lane & 15;
  const int hf   = lane >> 4;          // half-wave: 0 or 1
  const int wm   = wave >> 1;          // 0..3
  const int wn   = wave & 1;           // 0..1
  const long row0 = (long)blockIdx.y * 128 + (long)wm * 32;
  const long col0 = (long)blockIdx.x * 128 + (long)wn * 64;

  const __bf16* a0p = A + (row0 + ln) * (long)K;
  const __bf16* a1p = A + (row0 + 16 + ln) * (long)K;
  const __bf16* b0p = W + (col0 + ln) * (long)K;
  const __bf16* b1p = W + (col0 + 16 + ln) * (long)K;
  const __bf16* b2p = W + (col0 + 32 + ln) * (long)K;
  const __bf16* b3p = W + (col0 + 48 + ln) * (long)K;
  const int aoff = hf * 16;   // A lane K base within 32-block (contiguous, matches B)
  const int boff = hf * 16;   // B lane K base within 32-block (contiguous, matches A)

  v8f c00 = {}, c01 = {}, c02 = {}, c03 = {};
  v8f c10 = {}, c11 = {}, c12 = {}, c13 = {};
  for (int k = 0; k < K; k += 32) {
    v8bf a0lo = *(const v8bf*)(a0p + k + aoff);
    v8bf a0hi = *(const v8bf*)(a0p + k + 8 + aoff);
    v8bf a1lo = *(const v8bf*)(a1p + k + aoff);
    v8bf a1hi = *(const v8bf*)(a1p + k + 8 + aoff);
    v16bf a0 = __builtin_shufflevector(a0lo, a0hi, 0,1,2,3,4,5,6,7,8,9,10,11,12,13,14,15);
    v16bf a1 = __builtin_shufflevector(a1lo, a1hi, 0,1,2,3,4,5,6,7,8,9,10,11,12,13,14,15);
    v16bf b0 = *(const v16bf*)(b0p + k + boff);
    v16bf b1 = *(const v16bf*)(b1p + k + boff);
    v16bf b2 = *(const v16bf*)(b2p + k + boff);
    v16bf b3 = *(const v16bf*)(b3p + k + boff);
    c00 = __builtin_amdgcn_wmma_f32_16x16x32_bf16(false, a0, false, b0, (short)0, c00, false, false);
    c01 = __builtin_amdgcn_wmma_f32_16x16x32_bf16(false, a0, false, b1, (short)0, c01, false, false);
    c02 = __builtin_amdgcn_wmma_f32_16x16x32_bf16(false, a0, false, b2, (short)0, c02, false, false);
    c03 = __builtin_amdgcn_wmma_f32_16x16x32_bf16(false, a0, false, b3, (short)0, c03, false, false);
    c10 = __builtin_amdgcn_wmma_f32_16x16x32_bf16(false, a1, false, b0, (short)0, c10, false, false);
    c11 = __builtin_amdgcn_wmma_f32_16x16x32_bf16(false, a1, false, b1, (short)0, c11, false, false);
    c12 = __builtin_amdgcn_wmma_f32_16x16x32_bf16(false, a1, false, b2, (short)0, c12, false, false);
    c13 = __builtin_amdgcn_wmma_f32_16x16x32_bf16(false, a1, false, b3, (short)0, c13, false, false);
  }
  store_frag<MODE>(c00, row0,      col0,      hf, ln, bias, resid, outF, outB, (long)N);
  store_frag<MODE>(c01, row0,      col0 + 16, hf, ln, bias, resid, outF, outB, (long)N);
  store_frag<MODE>(c02, row0,      col0 + 32, hf, ln, bias, resid, outF, outB, (long)N);
  store_frag<MODE>(c03, row0,      col0 + 48, hf, ln, bias, resid, outF, outB, (long)N);
  store_frag<MODE>(c10, row0 + 16, col0,      hf, ln, bias, resid, outF, outB, (long)N);
  store_frag<MODE>(c11, row0 + 16, col0 + 16, hf, ln, bias, resid, outF, outB, (long)N);
  store_frag<MODE>(c12, row0 + 16, col0 + 32, hf, ln, bias, resid, outF, outB, (long)N);
  store_frag<MODE>(c13, row0 + 16, col0 + 48, hf, ln, bias, resid, outF, outB, (long)N);
}

// ---------------- m-axis attention (seq len 4) ----------------
// thread per (b,n,h,mq). qkv: [token, 2304], q/k/v at +0/+768/+1536, head offset h*64.
__global__ __launch_bounds__(256) void attn_m_kernel(const __bf16* __restrict__ qkv,
                                                     __bf16* __restrict__ xm) {
  const int gid = blockIdx.x * 256 + threadIdx.x;     // 256*49*12*4 = 602112 total
  const int mq = gid & 3;
  const int h  = (gid >> 2) % NH;
  const long bn = gid / (NH * 4);                     // b*49 + n
  const long tq = bn * 4 + mq;

  const __bf16* qp = qkv + tq * 2304 + h * HD;
  float qr[HD];
#pragma unroll
  for (int c = 0; c < HD; c++) qr[c] = bf2f(qp[c]);

  float s[4];
#pragma unroll
  for (int p = 0; p < 4; p++) {
    const __bf16* kp = qkv + (bn * 4 + p) * 2304 + 768 + h * HD;
    float acc = 0.f;
#pragma unroll
    for (int c = 0; c < HD; c++) acc += qr[c] * bf2f(kp[c]);
    s[p] = acc * 0.125f;
  }
  float mx = fmaxf(fmaxf(s[0], s[1]), fmaxf(s[2], s[3]));
  float sum = 0.f;
#pragma unroll
  for (int p = 0; p < 4; p++) { s[p] = __expf(s[p] - mx); sum += s[p]; }
  const float inv = 1.0f / sum;
#pragma unroll
  for (int p = 0; p < 4; p++) s[p] *= inv;

  __bf16* op = xm + tq * CDIM + h * HD;
  for (int c = 0; c < HD; c++) {
    float acc = 0.f;
#pragma unroll
    for (int p = 0; p < 4; p++)
      acc += s[p] * bf2f(qkv[(bn * 4 + p) * 2304 + 1536 + h * HD + c]);
    op[c] = f2bf(acc);
  }
}

// ---------------- n-axis attention (seq len 49) ----------------
// block per (b,m,h). K staged in LDS (later reused for V); scores in LDS.
__global__ __launch_bounds__(256) void attn_n_kernel(const __bf16* __restrict__ qkv,
                                                     __bf16* __restrict__ xn) {
  __shared__ float kb[49 * HD];   // K, later reused for V
  __shared__ float sb[49 * 49];   // scores
  const int unit = blockIdx.x;    // 256*4*12 = 12288
  const int h = unit % NH;
  const int m = (unit / NH) & 3;
  const int b = unit / (NH * 4);
  const int tid = threadIdx.x;

  // token for key/query row n: ((b*49+n)*4 + m)
  for (int idx = tid; idx < 49 * HD; idx += 256) {
    const int n = idx >> 6, c = idx & 63;
    const long t = ((long)(b * 49 + n)) * 4 + m;
    kb[idx] = bf2f(qkv[t * 2304 + 768 + h * HD + c]);
  }
  __syncthreads();

  // scores: q loaded from global (L2-resident), K from LDS
  for (int idx = tid; idx < 49 * 49; idx += 256) {
    const int qi = idx / 49, p = idx % 49;
    const long tq = ((long)(b * 49 + qi)) * 4 + m;
    const __bf16* qp = qkv + tq * 2304 + h * HD;
    float acc = 0.f;
#pragma unroll
    for (int c = 0; c < HD; c++) acc += bf2f(qp[c]) * kb[p * HD + c];
    sb[idx] = acc * 0.125f;
  }
  __syncthreads();

  // load V over kb (independent of softmax on sb)
  for (int idx = tid; idx < 49 * HD; idx += 256) {
    const int n = idx >> 6, c = idx & 63;
    const long t = ((long)(b * 49 + n)) * 4 + m;
    kb[idx] = bf2f(qkv[t * 2304 + 1536 + h * HD + c]);
  }
  // softmax rows of sb
  for (int row = tid; row < 49; row += 256) {
    float mx = -1e30f;
    for (int p = 0; p < 49; p++) mx = fmaxf(mx, sb[row * 49 + p]);
    float sum = 0.f;
    for (int p = 0; p < 49; p++) { float e = __expf(sb[row * 49 + p] - mx); sb[row * 49 + p] = e; sum += e; }
    const float inv = 1.0f / sum;
    for (int p = 0; p < 49; p++) sb[row * 49 + p] *= inv;
  }
  __syncthreads();

  // output: A·V
  for (int idx = tid; idx < 49 * HD; idx += 256) {
    const int qi = idx >> 6, c = idx & 63;
    float acc = 0.f;
    for (int p = 0; p < 49; p++) acc += sb[qi * 49 + p] * kb[p * HD + c];
    const long tq = ((long)(b * 49 + qi)) * 4 + m;
    xn[tq * CDIM + h * HD + c] = f2bf(acc);
  }
}

// ---------------- mix combine: x_ = xn*w0 + xm*w1, softmax(mix) over last dim ----------------
__global__ __launch_bounds__(256) void mix_kernel(const __bf16* __restrict__ xn,
                                                  const __bf16* __restrict__ xm,
                                                  const float* __restrict__ mix,
                                                  __bf16* __restrict__ out) {
  const long i = (long)blockIdx.x * 256 + threadIdx.x;  // TOK*768 total (exact)
  const long t = i / CDIM;
  const int m = (int)(t & 3);
  const int n = (int)((t >> 2) % 49);
  const float l0 = mix[(n * 4 + m) * 2 + 0];
  const float l1 = mix[(n * 4 + m) * 2 + 1];
  const float mx = fmaxf(l0, l1);
  const float e0 = __expf(l0 - mx), e1 = __expf(l1 - mx);
  const float inv = 1.0f / (e0 + e1);
  out[i] = f2bf(bf2f(xn[i]) * (e0 * inv) + bf2f(xm[i]) * (e1 * inv));
}

// ---------------- host launch ----------------
extern "C" void kernel_launch(void* const* d_in, const int* in_sizes, int n_in,
                              void* d_out, int out_size, void* d_ws, size_t ws_size,
                              hipStream_t stream) {
  const float* x     = (const float*)d_in[0];
  const float* g1    = (const float*)d_in[1];
  const float* bb1   = (const float*)d_in[2];
  const float* Wqkv  = (const float*)d_in[3];
  const float* Wproj = (const float*)d_in[4];
  const float* bproj = (const float*)d_in[5];
  const float* mixp  = (const float*)d_in[6];
  const float* g2    = (const float*)d_in[7];
  const float* bb2   = (const float*)d_in[8];
  const float* Wfc1  = (const float*)d_in[9];
  const float* bfc1  = (const float*)d_in[10];
  const float* Wfc2  = (const float*)d_in[11];
  const float* bfc2  = (const float*)d_in[12];
  float* out = (float*)d_out;

  size_t off = 0;
  char* base = (char*)d_ws;
  auto take = [&](size_t bytes) -> char* {
    char* p = base + off;
    off += (bytes + 255) & ~(size_t)255;
    return p;
  };
  __bf16* wq   = (__bf16*)take(2304L * 768 * 2);
  __bf16* wp   = (__bf16*)take(768L * 768 * 2);
  __bf16* w1   = (__bf16*)take(3072L * 768 * 2);
  __bf16* w2   = (__bf16*)take(768L * 3072 * 2);
  __bf16* Ab   = (__bf16*)take(TOK * 768 * 2);      // ln1 out, reused for ln2 out
  char*   R    = take(TOK * 3072 * 2);              // qkv + xm packed; later aliased by h
  __bf16* qkv  = (__bf16*)R;                        // TOK*2304
  __bf16* xm   = (__bf16*)(R + TOK * 2304 * 2);     // TOK*768
  __bf16* hbuf = (__bf16*)R;                        // TOK*3072 (after qkv/xm are dead)
  __bf16* xn   = (__bf16*)take(TOK * 768 * 2);
  __bf16* xmix = (__bf16*)take(TOK * 768 * 2);

  // weight converts
  cvt_bf16_kernel<<<(2304L * 768 + 255) / 256, 256, 0, stream>>>(Wqkv, wq, 2304L * 768);
  cvt_bf16_kernel<<<(768L * 768 + 255) / 256, 256, 0, stream>>>(Wproj, wp, 768L * 768);
  cvt_bf16_kernel<<<(3072L * 768 + 255) / 256, 256, 0, stream>>>(Wfc1, w1, 3072L * 768);
  cvt_bf16_kernel<<<(768L * 3072 + 255) / 256, 256, 0, stream>>>(Wfc2, w2, 768L * 3072);

  // LN1 -> bf16
  ln_bf16_kernel<<<(int)TOK, 256, 0, stream>>>(x, g1, bb1, Ab);

  // QKV GEMM: [TOK,768] x [2304,768]^T -> bf16 qkv
  gemm_wmma_kernel<0><<<dim3(2304 / 128, TOK / 128), 256, 0, stream>>>(
      Ab, wq, nullptr, nullptr, nullptr, qkv, 2304, 768);

  // axial attentions
  attn_m_kernel<<<(256 * 49 * 12 * 4) / 256, 256, 0, stream>>>(qkv, xm);
  attn_n_kernel<<<256 * 4 * 12, 256, 0, stream>>>(qkv, xn);

  // mix combine -> bf16
  mix_kernel<<<(int)((TOK * 768) / 256), 256, 0, stream>>>(xn, xm, mixp, xmix);

  // proj GEMM + bias + residual(x) -> out (= x_new, fp32)
  gemm_wmma_kernel<1><<<dim3(768 / 128, TOK / 128), 256, 0, stream>>>(
      xmix, wp, bproj, x, out, nullptr, 768, 768);

  // LN2 on x_new -> bf16
  ln_bf16_kernel<<<(int)TOK, 256, 0, stream>>>(out, g2, bb2, Ab);

  // FC1 GEMM + bias + GELU -> bf16 h
  gemm_wmma_kernel<2><<<dim3(3072 / 128, TOK / 128), 256, 0, stream>>>(
      Ab, w1, bfc1, nullptr, nullptr, hbuf, 3072, 768);

  // FC2 GEMM + bias + residual(x_new) -> final out
  gemm_wmma_kernel<1><<<dim3(768 / 128, TOK / 128), 256, 0, stream>>>(
      hbuf, w2, bfc2, out, out, nullptr, 768, 3072);
}
